// NodeModel_19851338842522
// MI455X (gfx1250) — matrix operand order repaired and checked by reading, not compiled, not run on hardware
//
#include <hip/hip_runtime.h>

#define N_NODES 100000
#define N_EDGES 1600000
#define D_IN    272
#define NT      8      // 128/16 output-column tiles
#define K1      9      // ceil(288/32) k-steps for layer 1 (K padded 272->288)
#define K2      4      // 128/32 k-steps for layers 2/3
#define M_TILES (N_NODES / 16)   // 6250, exact
#define LN_EPS  1e-5f

typedef __attribute__((ext_vector_type(16))) __bf16 v16bf;
typedef __attribute__((ext_vector_type(2)))  __bf16 v2bf;
typedef __attribute__((ext_vector_type(8)))  float  v8f;

// pack two f32 -> two bf16 in one dword (lo in [15:0], hi in [31:16]).
// round-half-up + v_perm_b32: 3 VALU ops per pair.
static __device__ __forceinline__ unsigned f2bf2(float lo, float hi) {
  unsigned ulo = __float_as_uint(lo) + 0x8000u;
  unsigned uhi = __float_as_uint(hi) + 0x8000u;
  return __builtin_amdgcn_perm(uhi, ulo, 0x07060302u);
}

// native-cast variant (cold path probe: does gfx1250 have a hw f32->bf16 cvt?)
static __device__ __forceinline__ unsigned f2bf2_native(float lo, float hi) {
  v2bf v;
  v[0] = (__bf16)lo;
  v[1] = (__bf16)hi;
  unsigned r;
  __builtin_memcpy(&r, &v, 4);
  return r;
}

static __device__ __forceinline__ v8f wmma_bf16(v16bf a, v16bf b, v8f c) {
  // (neg_a, A, neg_b, B, c_mod, C, reuse_a, reuse_b)
  return __builtin_amdgcn_wmma_f32_16x16x32_bf16(false, a, false, b, (short)0, c,
                                                 false, false);
}

// ---------------------------------------------------------------- zero agg
__global__ void zero_agg(float4* __restrict__ agg) {
  int i = blockIdx.x * blockDim.x + threadIdx.x;
  if (i < N_NODES * 128 / 4) agg[i] = make_float4(0.f, 0.f, 0.f, 0.f);
}

// ------------------------------------------------- scatter-add (segment_sum)
__global__ void scatter_edges(const float* __restrict__ edge_attr,
                              const int* __restrict__ edge_index,
                              float* __restrict__ agg) {
  int gid = blockIdx.x * blockDim.x + threadIdx.x;   // N_EDGES*32 = 51.2M
  if (gid >= N_EDGES * 32) return;
  int e = gid >> 5, c = gid & 31;                    // 1 wave <-> 1 edge (coalesced)
  int dst = edge_index[N_EDGES + e];                 // row 1 = destinations
  float4 v = ((const float4*)edge_attr)[(size_t)e * 32 + c];
  float* p = agg + (size_t)dst * 128 + c * 4;
  atomicAdd(p + 0, v.x);
  atomicAdd(p + 1, v.y);
  atomicAdd(p + 2, v.z);
  atomicAdd(p + 3, v.w);
}

// --------------------------- pre-swizzle weights into bf16 B-fragment layout
// B frag (16x16x32): lane L holds N = L%16, element i holds K = 32*ks + 8*(L/16) + (i<8 ? i : i+8)
__global__ void prep_weights(const float* __restrict__ W1,
                             const float* __restrict__ W2,
                             const float* __restrict__ W3,
                             unsigned* __restrict__ W1f,
                             unsigned* __restrict__ W2f,
                             unsigned* __restrict__ W3f) {
  int g = blockIdx.x * blockDim.x + threadIdx.x;     // (frag, lane)
  if (g >= 136 * 32) return;
  int lane = g & 31, f = g >> 5;
  int half = lane >> 4, n16 = lane & 15;
  const float* W;
  unsigned* out;
  int fl, kdim;
  if (f < 72)        { W = W1; out = W1f; fl = f;       kdim = D_IN; }
  else if (f < 104)  { W = W2; out = W2f; fl = f - 72;  kdim = 128;  }
  else               { W = W3; out = W3f; fl = f - 104; kdim = 128;  }
  int kstep = fl >> 3, t = fl & 7;
  int n = n16 + 16 * t;
  unsigned vals[8];
#pragma unroll
  for (int i = 0; i < 16; i += 2) {
    int k0 = kstep * 32 + 8 * half + (i < 8 ? i : i + 8);
    float a = (k0 < kdim)     ? W[(size_t)k0 * 128 + n]       : 0.0f;
    float b = (k0 + 1 < kdim) ? W[(size_t)(k0 + 1) * 128 + n] : 0.0f;
    vals[i >> 1] = f2bf2_native(a, b);     // probe hw cvt in cold path
  }
  uint4* d4 = (uint4*)(out + ((size_t)(fl * 32 + lane)) * 8);
  d4[0] = ((const uint4*)vals)[0];
  d4[1] = ((const uint4*)vals)[1];
}

// ------------------------------------------------------------- fused 3-layer MLP
__global__ __launch_bounds__(128, 1) void mlp_fused(
    const float* __restrict__ x, const float* __restrict__ agg,
    const float* __restrict__ u,
    const unsigned* __restrict__ W1f, const float* __restrict__ b1,
    const float* __restrict__ g1, const float* __restrict__ be1,
    const unsigned* __restrict__ W2f, const float* __restrict__ b2,
    const float* __restrict__ g2, const float* __restrict__ be2,
    const unsigned* __restrict__ W3f, const float* __restrict__ b3,
    float* __restrict__ out) {
  __shared__ unsigned short act[4][16 * 128];        // bf16 staging, 4KB/wave

  const int lane = threadIdx.x & 31;
  const int wave = threadIdx.x >> 5;
  const int tile = blockIdx.x * 4 + wave;
  const bool active = tile < M_TILES;
  const int tilec = active ? tile : (M_TILES - 1);   // clamp: uniform control flow
  const int rowbase = tilec * 16;
  const int half = lane >> 4, lm = lane & 15;
  const int arow = rowbase + lm;                     // row this lane feeds into A

  // load 8 consecutive f32 of the concat [x|agg|u|0-pad], convert to 4 packed bf16 dwords
  auto load_run = [&](int k, unsigned* o4) {
    float4 v0, v1;
    if (k < 128) {
      const float4* p = (const float4*)(x + (size_t)arow * 128 + k);
      v0 = p[0]; v1 = p[1];
    } else if (k < 256) {
      const float4* p = (const float4*)(agg + (size_t)arow * 128 + (k - 128));
      v0 = p[0]; v1 = p[1];
    } else if (k < D_IN) {
      const float4* p = (const float4*)(u + (k - 256));
      v0 = p[0]; v1 = p[1];
    } else {
      v0 = v1 = make_float4(0.f, 0.f, 0.f, 0.f);
    }
    o4[0] = f2bf2(v0.x, v0.y);
    o4[1] = f2bf2(v0.z, v0.w);
    o4[2] = f2bf2(v1.x, v1.y);
    o4[3] = f2bf2(v1.z, v1.w);
  };

  auto load_bfrag = [&](const unsigned* Wf, int fi) -> v16bf {
    const uint4* wp = (const uint4*)Wf;
    uint4 q0 = wp[fi * 2 + 0];
    uint4 q1 = wp[fi * 2 + 1];
    v16bf b;
    __builtin_memcpy(&b, &q0, 16);
    __builtin_memcpy((char*)&b + 16, &q1, 16);
    return b;
  };

  // A frag from bf16 LDS tile: lane L = row lm, element i: K = 32ks + 8*half + (i<8 ? i : i+8)
  auto a_from_lds = [&](int ks) -> v16bf {
    int kbase = ks * 32 + 8 * half;
    uint4 q0 = *(const uint4*)&act[wave][lm * 128 + kbase];
    uint4 q1 = *(const uint4*)&act[wave][lm * 128 + kbase + 16];
    v16bf a;
    __builtin_memcpy(&a, &q0, 16);
    __builtin_memcpy((char*)&a + 16, &q1, 16);
    return a;
  };

  // bias + relu + layernorm, in C layout (VGPR r -> rows r / r+8; lane -> col)
  auto post = [&](v8f* a, const float* bias, const float* gam, const float* bet) {
#pragma unroll
    for (int t = 0; t < NT; ++t) {
      float bv = bias[lm + 16 * t];
#pragma unroll
      for (int r = 0; r < 8; ++r) {
        float v = a[t][r] + bv;
        a[t][r] = v > 0.f ? v : 0.f;
      }
    }
#pragma unroll
    for (int r = 0; r < 8; ++r) {
      float s = 0.f, q = 0.f;
#pragma unroll
      for (int t = 0; t < NT; ++t) { float v = a[t][r]; s += v; q += v * v; }
      for (int m = 1; m < 16; m <<= 1) {
        s += __shfl_xor(s, m, 16);
        q += __shfl_xor(q, m, 16);
      }
      float mean = s * (1.f / 128.f);
      float var  = q * (1.f / 128.f) - mean * mean;
      float rstd = rsqrtf(var + LN_EPS);
#pragma unroll
      for (int t = 0; t < NT; ++t) {
        int col = lm + 16 * t;
        a[t][r] = (a[t][r] - mean) * rstd * gam[col] + bet[col];
      }
    }
  };

  // write C tile to LDS as bf16, converting pairs (t, t+1) with one packed cvt
  auto to_lds = [&](v8f* a) {
#pragma unroll
    for (int t = 0; t < NT; t += 2)
#pragma unroll
      for (int r = 0; r < 8; ++r) {
        int m = r + 8 * half;
        unsigned p = f2bf2(a[t][r], a[t + 1][r]);
        act[wave][m * 128 + lm + 16 * t]        = (unsigned short)p;
        act[wave][m * 128 + lm + 16 * (t + 1)]  = (unsigned short)(p >> 16);
      }
  };

  // ---------------- layer 1: [x|agg|u] @ W1  (K = 288 padded) ----------------
  v8f acc[NT];
#pragma unroll
  for (int t = 0; t < NT; ++t)
#pragma unroll
    for (int i = 0; i < 8; ++i) acc[t][i] = 0.f;

  for (int ks = 0; ks < K1; ++ks) {
    unsigned af[8];
    int kbase = ks * 32 + 8 * half;
    load_run(kbase, af);
    load_run(kbase + 16, af + 4);
    v16bf a;
    __builtin_memcpy(&a, af, 32);
    v16bf bfr[NT];
#pragma unroll
    for (int t = 0; t < NT; ++t)
      bfr[t] = load_bfrag(W1f, (ks * NT + t) * 32 + lane);
#pragma unroll
    for (int t = 0; t < NT; ++t)
      acc[t] = wmma_bf16(a, bfr[t], acc[t]);
  }
  post(acc, b1, g1, be1);
  to_lds(acc);
  __syncthreads();

  // ---------------- layer 2 ----------------
  v8f acc2[NT];
#pragma unroll
  for (int t = 0; t < NT; ++t)
#pragma unroll
    for (int i = 0; i < 8; ++i) acc2[t][i] = 0.f;

  for (int ks = 0; ks < K2; ++ks) {
    v16bf a = a_from_lds(ks);
    v16bf bfr[NT];
#pragma unroll
    for (int t = 0; t < NT; ++t)
      bfr[t] = load_bfrag(W2f, (ks * NT + t) * 32 + lane);
#pragma unroll
    for (int t = 0; t < NT; ++t)
      acc2[t] = wmma_bf16(a, bfr[t], acc2[t]);
  }
  post(acc2, b2, g2, be2);
  __syncthreads();
  to_lds(acc2);
  __syncthreads();

  // ---------------- layer 3 (no relu / LN) ----------------
  v8f acc3[NT];
#pragma unroll
  for (int t = 0; t < NT; ++t)
#pragma unroll
    for (int i = 0; i < 8; ++i) acc3[t][i] = 0.f;

  for (int ks = 0; ks < K2; ++ks) {
    v16bf a = a_from_lds(ks);
    v16bf bfr[NT];
#pragma unroll
    for (int t = 0; t < NT; ++t)
      bfr[t] = load_bfrag(W3f, (ks * NT + t) * 32 + lane);
#pragma unroll
    for (int t = 0; t < NT; ++t)
      acc3[t] = wmma_bf16(a, bfr[t], acc3[t]);
  }

  if (active) {
#pragma unroll
    for (int t = 0; t < NT; ++t) {
      float bv = b3[lm + 16 * t];
#pragma unroll
      for (int r = 0; r < 8; ++r) {
        int row = rowbase + r + 8 * half;
        out[(size_t)row * 128 + lm + 16 * t] = acc3[t][r] + bv;
      }
    }
  }
}

// ---------------------------------------------------------------------------
extern "C" void kernel_launch(void* const* d_in, const int* in_sizes, int n_in,
                              void* d_out, int out_size, void* d_ws, size_t ws_size,
                              hipStream_t stream) {
  const float* x   = (const float*)d_in[0];
  const float* ea  = (const float*)d_in[1];
  const float* u   = (const float*)d_in[2];
  const float* W1  = (const float*)d_in[3];
  const float* b1  = (const float*)d_in[4];
  const float* g1  = (const float*)d_in[5];
  const float* be1 = (const float*)d_in[6];
  const float* W2  = (const float*)d_in[7];
  const float* b2  = (const float*)d_in[8];
  const float* g2  = (const float*)d_in[9];
  const float* be2 = (const float*)d_in[10];
  const float* W3  = (const float*)d_in[11];
  const float* b3  = (const float*)d_in[12];
  const int*   ei  = (const int*)d_in[13];
  float* out = (float*)d_out;

  char* ws = (char*)d_ws;
  float* agg = (float*)ws;                                   // 51.2 MB
  size_t o = (size_t)N_NODES * 128 * sizeof(float);
  o = (o + 255) & ~(size_t)255;
  unsigned* W1f = (unsigned*)(ws + o);                       // 72 frags
  o += (size_t)72 * 512 * 2;  o = (o + 255) & ~(size_t)255;
  unsigned* W2f = (unsigned*)(ws + o);                       // 32 frags
  o += (size_t)32 * 512 * 2;  o = (o + 255) & ~(size_t)255;
  unsigned* W3f = (unsigned*)(ws + o);                       // 32 frags

  zero_agg<<<(N_NODES * 128 / 4 + 255) / 256, 256, 0, stream>>>((float4*)agg);
  prep_weights<<<(136 * 32 + 127) / 128, 128, 0, stream>>>(W1, W2, W3, W1f, W2f, W3f);
  scatter_edges<<<(N_EDGES * 32) / 256, 256, 0, stream>>>(ea, ei, agg);
  mlp_fused<<<(M_TILES + 3) / 4, 128, 0, stream>>>(x, agg, u,
                                                   W1f, b1, g1, be1,
                                                   W2f, b2, g2, be2,
                                                   W3f, b3, out);
}
